// PhasedSNForecastModel_42606075576840
// MI455X (gfx1250) — matrix operand order repaired and checked by reading, not compiled, not run on hardware
//
#include <hip/hip_runtime.h>
#include <hip/hip_bf16.h>

typedef __attribute__((ext_vector_type(16))) __bf16 v16bf;
typedef __attribute__((ext_vector_type(8)))  float  v8f;
typedef __attribute__((ext_vector_type(4)))  unsigned int v4u;
typedef __attribute__((ext_vector_type(8)))  int  v8i_t;
typedef __attribute__((ext_vector_type(4)))  int  v4i_t;

#define B_TOT   1024
#define SEQ     512
#define FEAT    2
#define U1N     128
#define U2N     64
#define UDN     32
#define OUTS    32
#define ALPHA_K 0.001f

// packed-weight u16 offsets (identical in workspace and the LDS mirror)
#define OFF_U1  0        // 4 ktile * 32 ntile
#define OFF_W1  65536    // 1 ktile * 32 ntile  (x pad tile: K rows 0..1 = W1, rest 0)
#define OFF_W2  81920    // 4 ktile * 16 ntile
#define OFF_U2  114688   // 2 ktile * 16 ntile
#define OFF_WD  131072   // 2 ktile * 2 ntile
#define OFF_WO  133120   // 1 ktile * 1 ntile (cols >= 2 zero)
#define W_U16_TOTAL 133632           // 267,264 bytes = 66,816 dwords = 8352 x 8

// LDS byte offsets
#define LDS_W    0
#define LDS_H1   267264   // 16 x 128 bf16
#define LDS_H2   271360   // 16 x 64 bf16
#define LDS_D    273408   // 16 x 32 bf16
#define LDS_X    274432   // 16 x 2 f32
#define LDS_MIN  274560   // 16 x 2 f32
#define LDS_RNG  274688   // 16 x 2 f32
#define LDS_RMIN 274816   // 4 x 32 f32
#define LDS_RMAX 275328   // 4 x 32 f32
#define LDS_TOTAL 275840

// TDM tile geometry for the weight stage: 8352 dwords x 8 rows == 267,264 B
#define TDM_ROW_DW 8352
#define TDM_NROWS  8

static __device__ __forceinline__ unsigned short f2bf_u16(float f) {
    unsigned u = __float_as_uint(f);
    u += 0x7fffu + ((u >> 16) & 1u);          // round-to-nearest-even
    return (unsigned short)(u >> 16);
}
static __device__ __forceinline__ __bf16 f2bf(float f) {
    return __builtin_bit_cast(__bf16, f2bf_u16(f));
}
static __device__ __forceinline__ float sigf(float x) {
    // v_mul+v_exp+v_add+v_rcp -- avoids the IEEE-div Newton sequence
    return __builtin_amdgcn_rcpf(1.0f + __expf(-x));
}
static __device__ __forceinline__ float kgate(float t, float tau, float itau,
                                              float sh, float ron, float iron) {
    float x   = t - sh;
    float phi = (x - tau * floorf(x * itau)) * itau;   // mod(t-shift,tau)/tau in [0,1)
    float k   = (phi < 0.5f * ron) ? (2.0f * phi * iron)
              : ((phi < ron)       ? (2.0f - 2.0f * phi * iron)
                                   : (ALPHA_K * phi));
    return k;
}

// ---------------------------------------------------------------------------
// Pack a row-major (K x N) f32 weight into WMMA-B-ready bf16 tiles:
//   dst[(kt*ntiles + nt)*32 + lane][kk] = W[kt*32 + (lane>>4)*16 + kk][nt*16 + lane&15]
// so each lane's 16 bf16 (32B) load directly feeds v_wmma_f32_16x16x32_bf16 SRC1.
// ---------------------------------------------------------------------------
__global__ void pack_b_tiles(const float* __restrict__ W, unsigned short* __restrict__ dst,
                             int K, int N, int ktiles, int ntiles) {
    int idx = blockIdx.x * blockDim.x + threadIdx.x;
    int total = ktiles * ntiles * 32;
    if (idx >= total) return;
    int lane = idx & 31;
    int tile = idx >> 5;
    int nt = tile % ntiles;
    int kt = tile / ntiles;
    int n = nt * 16 + (lane & 15);
    int kbase = kt * 32 + (lane >> 4) * 16;
    unsigned short* d = dst + (size_t)idx * 16;
#pragma unroll
    for (int kk = 0; kk < 16; ++kk) {
        int k = kbase + kk;
        float v = (k < K && n < N) ? W[(size_t)k * N + n] : 0.0f;
        d[kk] = f2bf_u16(v);
    }
}

// ---------------------------------------------------------------------------
// Main kernel: one workgroup (4 waves) per 16-row batch tile. Runs the whole
// 512-step warm scan plus the 32-step rollout with LDS-resident bf16 weights
// (staged by the Tensor Data Mover where available).
// ---------------------------------------------------------------------------
__global__ __launch_bounds__(128, 1)
void plstm_forecast_kernel(const float* __restrict__ inputs,
                           const float* __restrict__ b1,
                           const float* __restrict__ tau1, const float* __restrict__ shift1,
                           const float* __restrict__ ron1,
                           const float* __restrict__ b2,
                           const float* __restrict__ tau2, const float* __restrict__ shift2,
                           const float* __restrict__ ron2,
                           const float* __restrict__ bd1, const float* __restrict__ bo,
                           const unsigned short* __restrict__ wpk,
                           float* __restrict__ out)
{
    extern __shared__ char smem[];
    unsigned short* wl  = (unsigned short*)(smem + LDS_W);
    unsigned short* h1u = (unsigned short*)(smem + LDS_H1);
    unsigned short* h2u = (unsigned short*)(smem + LDS_H2);
    unsigned short* du  = (unsigned short*)(smem + LDS_D);
    float* xl  = (float*)(smem + LDS_X);
    float* mnl = (float*)(smem + LDS_MIN);
    float* rgl = (float*)(smem + LDS_RNG);
    float* rmn = (float*)(smem + LDS_RMIN);
    float* rmx = (float*)(smem + LDS_RMAX);

    const int tid  = threadIdx.x;
    const int w    = tid >> 5;      // wave id 0..3
    const int lane = tid & 31;
    const int col  = lane & 15;     // N column within a 16-wide tile
    const int lhi  = lane >> 4;     // lane half (selects M 0..7 vs 8..15, K halves)
    const int b0   = blockIdx.x * 16;

    // ---- stage packed weights (global -> LDS) ----
#if __has_builtin(__builtin_amdgcn_tensor_load_to_lds)
    if (w == 0) {
        // Tensor Data Mover: one 2D descriptor moves all 267,264 weight bytes.
        // D# group0: count=1 | lds_addr=0 | global_addr | type=2 ("image")
        unsigned long long ga = (unsigned long long)wpk;
        v4u g0 = { 1u, 0u,
                   (unsigned)(ga & 0xffffffffu),
                   (unsigned)(((ga >> 32) & 0x01ffffffu) | 0x80000000u) };
        // D# group1: data_size=4B; tensor_dim0=tile_dim0=stride0=8352 dw;
        //            tensor_dim1=tile_dim1=8; no pad/iterate/multicast.
        v8i_t g1 = { (int)0x00020000u,                     // data_size=2 (4B)
                     (int)((TDM_ROW_DW & 0xffff) << 16),   // tensor_dim0[15:0]
                     (int)(TDM_NROWS << 16),               // tensor_dim1[15:0]
                     (int)((TDM_ROW_DW & 0xffff) << 16),   // tile_dim0
                     (int)TDM_NROWS,                       // tile_dim1
                     (int)TDM_ROW_DW,                      // tensor_dim0_stride[31:0]
                     0, 0 };
        v4i_t gz4 = { 0, 0, 0, 0 };                        // 2D: groups 2/3 unused
        v8i_t gz8 = { 0, 0, 0, 0, 0, 0, 0, 0 };            // unused trailing group
        __builtin_amdgcn_tensor_load_to_lds(g0, g1, gz4, gz4, gz8, 0);
        __builtin_amdgcn_s_wait_tensorcnt(0);
    }
#else
    {
        const uint4* src = (const uint4*)wpk;
        uint4* dst = (uint4*)wl;
        for (int i = tid; i < (W_U16_TOTAL / 8); i += 128) dst[i] = src[i];
    }
#endif
    for (int i = tid; i < 16 * U1N; i += 128) h1u[i] = 0;
    for (int i = tid; i < 16 * U2N; i += 128) h2u[i] = 0;
    if (tid < 32) xl[tid] = 0.0f;

    // ---- per (row,feature) min / range over the sequence ----
    {
        int p = tid & 31;           // (row,feat) pair: 16 rows * 2 feats
        int g = tid >> 5;           // 4 groups split S=512
        int row = p >> 1, f = p & 1;
        float mn = 3.4e38f, mx = -3.4e38f;
        const float* base = inputs + (size_t)(b0 + row) * SEQ * FEAT + f;
        for (int s = g * 128; s < (g + 1) * 128; ++s) {
            float v = base[(size_t)s * FEAT];
            mn = fminf(mn, v); mx = fmaxf(mx, v);
        }
        rmn[g * 32 + p] = mn; rmx[g * 32 + p] = mx;
    }
    __syncthreads();
    if (tid < 32) {
        float mn = fminf(fminf(rmn[tid], rmn[32 + tid]), fminf(rmn[64 + tid], rmn[96 + tid]));
        float mx = fmaxf(fmaxf(rmx[tid], rmx[32 + tid]), fmaxf(rmx[64 + tid], rmx[96 + tid]));
        mnl[tid] = mn; rgl[tid] = mx - mn;
    }
    __syncthreads();

    // per-lane normalization registers (rows 0..15 <-> col; reciprocal ranges)
    const float xm0 = mnl[col * 2 + 0], xm1 = mnl[col * 2 + 1];
    const float xi0 = __builtin_amdgcn_rcpf(rgl[col * 2 + 0]);
    const float xi1 = __builtin_amdgcn_rcpf(rgl[col * 2 + 1]);

    // ---- per-lane recurrent parameters (wave w owns L1 units [32w,32w+32),
    //      L2 units [16w,16w+16); lane column selects the unit) ----
    float tv1[2], it1[2], sh1[2], rn1[2], ir1[2], bb1[4][2];
#pragma unroll
    for (int s = 0; s < 2; ++s) {
        int j = 32 * w + 16 * s + col;
        tv1[s] = tau1[j];  it1[s] = __builtin_amdgcn_rcpf(tv1[s]);
        sh1[s] = shift1[j];
        rn1[s] = ron1[j];  ir1[s] = __builtin_amdgcn_rcpf(rn1[s]);
#pragma unroll
        for (int q = 0; q < 4; ++q) bb1[q][s] = b1[q * U1N + j];
    }
    const int j2 = 16 * w + col;
    const float tv2 = tau2[j2], it2 = __builtin_amdgcn_rcpf(tv2), sh2v = shift2[j2];
    const float rn2v = ron2[j2], ir2 = __builtin_amdgcn_rcpf(rn2v);
    float bb2[4];
#pragma unroll
    for (int q = 0; q < 4; ++q) bb2[q] = b2[q * U2N + j2];
    const float bdv0 = bd1[col], bdv1 = bd1[16 + col];
    const float bov  = (col < FEAT) ? bo[col] : 0.0f;

    // ---- states in registers (WMMA C/D layout: vgpr v, lane -> M=v+8*lhi, N=col) ----
    v8f c1s[2], h1s[2], c2s, h2s;
#pragma unroll
    for (int v = 0; v < 8; ++v) {
        c1s[0][v] = 0.f; c1s[1][v] = 0.f; h1s[0][v] = 0.f; h1s[1][v] = 0.f;
        c2s[v] = 0.f; h2s[v] = 0.f;
    }

    // A-tile gather from row-major bf16 [16][width] LDS state (two b128 loads/lane)
    auto ldA = [&](const unsigned short* hp, int width, int kt) -> v16bf {
        int c0 = kt * 32 + 16 * lhi;
        v16bf a;
        uint4* ap = (uint4*)&a;
        ap[0] = *(const uint4*)(hp + col * width + c0);
        ap[1] = *(const uint4*)(hp + col * width + c0 + 8);
        return a;
    };
    auto ldB = [&](int off_u16, int tile) -> v16bf {
        return *(const v16bf*)(wl + off_u16 + (tile * 32 + lane) * 16);
    };
    auto mm = [&](v16bf a, v16bf b, v8f c) -> v8f {
        return __builtin_amdgcn_wmma_f32_16x16x32_bf16(false, a, false, b, (short)0, c,
                                                       false, false);
    };

    // ---- one PLSTM time step (both layers) ----
    auto step = [&](int t, bool warm) {
        const float tf = (float)t;
        // layer-1 A tiles: h1 state (K=128) + padded x tile (K=2 live)
        v16bf a1[4];
#pragma unroll
        for (int kt = 0; kt < 4; ++kt) a1[kt] = ldA(h1u, U1N, kt);
        v16bf ax;
#pragma unroll
        for (int e = 0; e < 16; ++e) ax[e] = __builtin_bit_cast(__bf16, (unsigned short)0);
        if (lane < 16) {
            float x0, x1;
            if (warm) {
                const float* gp = inputs + (size_t)(b0 + lane) * SEQ * FEAT + (size_t)t * FEAT;
                x0 = (gp[0] - xm0) * xi0;
                x1 = (gp[1] - xm1) * xi1;
            } else {
                x0 = xl[lane * 2 + 0];
                x1 = xl[lane * 2 + 1];
            }
            ax[0] = f2bf(x0); ax[1] = f2bf(x1);
        }
        // z1 tiles: gate q, subtile s -> N-tile 8q + 2w + s
        v8f acc1[4][2];
#pragma unroll
        for (int q = 0; q < 4; ++q)
#pragma unroll
            for (int s = 0; s < 2; ++s) {
                int nt = 8 * q + 2 * w + s;
                v8f c = {0.f, 0.f, 0.f, 0.f, 0.f, 0.f, 0.f, 0.f};
                c = mm(a1[0], ldB(OFF_U1, 0 * 32 + nt), c);
                c = mm(a1[1], ldB(OFF_U1, 1 * 32 + nt), c);
                c = mm(a1[2], ldB(OFF_U1, 2 * 32 + nt), c);
                c = mm(a1[3], ldB(OFF_U1, 3 * 32 + nt), c);
                c = mm(ax,    ldB(OFF_W1, nt),          c);
                acc1[q][s] = c;
            }
        // elementwise phased-LSTM update, layer 1
        unsigned short nh1b[2][8];
#pragma unroll
        for (int s = 0; s < 2; ++s) {
            float kv = kgate(tf, tv1[s], it1[s], sh1[s], rn1[s], ir1[s]);
#pragma unroll
            for (int v = 0; v < 8; ++v) {
                float iv = sigf(acc1[0][s][v] + bb1[0][s]);
                float fv = sigf(acc1[1][s][v] + bb1[1][s]);
                float gv = sigf(acc1[2][s][v] + bb1[2][s]);
                float ov = sigf(acc1[3][s][v] + bb1[3][s]);
                float ct = fv * c1s[s][v] + iv * gv;
                float ht = ov * sigf(ct);
                c1s[s][v] = c1s[s][v] + kv * (ct - c1s[s][v]);
                float hn  = h1s[s][v] + kv * (ht - h1s[s][v]);
                h1s[s][v] = hn;
                nh1b[s][v] = f2bf_u16(hn);
            }
        }
        __syncthreads();                       // all h1 reads complete
#pragma unroll
        for (int s = 0; s < 2; ++s) {
            int unit = 32 * w + 16 * s + col;
#pragma unroll
            for (int v = 0; v < 8; ++v)
                h1u[(v + 8 * lhi) * U1N + unit] = nh1b[s][v];
        }
        __syncthreads();                       // nh1 published

        // layer 2: z2 = nh1@W2 + h2@U2 ; gate q -> N-tile 4q + w
        v16bf a2[4], ah2[2];
#pragma unroll
        for (int kt = 0; kt < 4; ++kt) a2[kt] = ldA(h1u, U1N, kt);
#pragma unroll
        for (int kt = 0; kt < 2; ++kt) ah2[kt] = ldA(h2u, U2N, kt);
        v8f acc2[4];
#pragma unroll
        for (int q = 0; q < 4; ++q) {
            int nt = 4 * q + w;
            v8f c = {0.f, 0.f, 0.f, 0.f, 0.f, 0.f, 0.f, 0.f};
            c = mm(a2[0],  ldB(OFF_W2, 0 * 16 + nt), c);
            c = mm(a2[1],  ldB(OFF_W2, 1 * 16 + nt), c);
            c = mm(a2[2],  ldB(OFF_W2, 2 * 16 + nt), c);
            c = mm(a2[3],  ldB(OFF_W2, 3 * 16 + nt), c);
            c = mm(ah2[0], ldB(OFF_U2, 0 * 16 + nt), c);
            c = mm(ah2[1], ldB(OFF_U2, 1 * 16 + nt), c);
            acc2[q] = c;
        }
        unsigned short nh2b[8];
        {
            float kv = kgate(tf, tv2, it2, sh2v, rn2v, ir2);
#pragma unroll
            for (int v = 0; v < 8; ++v) {
                float iv = sigf(acc2[0][v] + bb2[0]);
                float fv = sigf(acc2[1][v] + bb2[1]);
                float gv = sigf(acc2[2][v] + bb2[2]);
                float ov = sigf(acc2[3][v] + bb2[3]);
                float ct = fv * c2s[v] + iv * gv;
                float ht = ov * sigf(ct);
                c2s[v] = c2s[v] + kv * (ct - c2s[v]);
                float hn = h2s[v] + kv * (ht - h2s[v]);
                h2s[v] = hn;
                nh2b[v] = f2bf_u16(hn);
            }
        }
        __syncthreads();                       // all h2 reads complete
        {
            int unit = 16 * w + col;
#pragma unroll
            for (int v = 0; v < 8; ++v)
                h2u[(v + 8 * lhi) * U2N + unit] = nh2b[v];
        }
        __syncthreads();                       // nh2 published
    };

    // ---- dense head: pred = sig(sig(h2@Wd1+bd1)@Wo+bo); store + feed back ----
    auto dense_store = [&](int r) {
        v16bf aA = ldA(h2u, U2N, 0);
        v16bf aB = ldA(h2u, U2N, 1);
#pragma unroll
        for (int nt = 0; nt < 2; ++nt) {
            v8f c = {0.f, 0.f, 0.f, 0.f, 0.f, 0.f, 0.f, 0.f};
            c = mm(aA, ldB(OFF_WD, 0 * 2 + nt), c);
            c = mm(aB, ldB(OFF_WD, 1 * 2 + nt), c);
            float bias = nt ? bdv1 : bdv0;
#pragma unroll
            for (int v = 0; v < 8; ++v)
                du[(v + 8 * lhi) * UDN + nt * 16 + col] = f2bf_u16(sigf(c[v] + bias));
        }
        v16bf ad = ldA(du, UDN, 0);            // same-wave LDS RAW (dscnt wait)
        v8f c = {0.f, 0.f, 0.f, 0.f, 0.f, 0.f, 0.f, 0.f};
        c = mm(ad, ldB(OFF_WO, 0), c);
        if (col < FEAT) {
#pragma unroll
            for (int v = 0; v < 8; ++v) {
                int row = v + 8 * lhi;
                float pv = sigf(c[v] + bov);
                xl[row * FEAT + col] = pv;     // normalized feedback input
                out[((size_t)(b0 + row) * OUTS + r) * FEAT + col] =
                    pv * rgl[row * FEAT + col] + mnl[row * FEAT + col];
            }
        }
    };

    // ---- warm scan over the observed sequence ----
#pragma unroll 1
    for (int t = 0; t < SEQ; ++t) step(t, true);

    // ---- autoregressive rollout ----
#pragma unroll 1
    for (int r = 0; r < OUTS; ++r) {
        if (w == 0) dense_store(r);
        __syncthreads();                       // pred visible in xl
        if (r < OUTS - 1) step(SEQ + r, false);
    }
}

extern "C" void kernel_launch(void* const* d_in, const int* in_sizes, int n_in,
                              void* d_out, int out_size, void* d_ws, size_t ws_size,
                              hipStream_t stream) {
    const float* inputs = (const float*)d_in[0];
    const float* W1  = (const float*)d_in[1];
    const float* U1  = (const float*)d_in[2];
    const float* b1  = (const float*)d_in[3];
    const float* tau1   = (const float*)d_in[4];
    const float* shift1 = (const float*)d_in[5];
    const float* ron1   = (const float*)d_in[6];
    const float* W2  = (const float*)d_in[7];
    const float* U2  = (const float*)d_in[8];
    const float* b2  = (const float*)d_in[9];
    const float* tau2   = (const float*)d_in[10];
    const float* shift2 = (const float*)d_in[11];
    const float* ron2   = (const float*)d_in[12];
    const float* Wd1 = (const float*)d_in[13];
    const float* bd1 = (const float*)d_in[14];
    const float* Wo  = (const float*)d_in[15];
    const float* bo  = (const float*)d_in[16];
    unsigned short* wpk = (unsigned short*)d_ws;
    float* out = (float*)d_out;

    auto pk = [&](const float* W, int off, int K, int N, int kt, int nt) {
        int total = kt * nt * 32;
        pack_b_tiles<<<(total + 127) / 128, 128, 0, stream>>>(W, wpk + off, K, N, kt, nt);
    };
    pk(U1,  OFF_U1, 128, 512, 4, 32);
    pk(W1,  OFF_W1,   2, 512, 1, 32);
    pk(W2,  OFF_W2, 128, 256, 4, 16);
    pk(U2,  OFF_U2,  64, 256, 2, 16);
    pk(Wd1, OFF_WD,  64,  32, 2,  2);
    pk(Wo,  OFF_WO,  32,   2, 1,  1);

    static int lds_cfg = [] {
        return (int)hipFuncSetAttribute((const void*)plstm_forecast_kernel,
                                        hipFuncAttributeMaxDynamicSharedMemorySize,
                                        LDS_TOTAL);
    }();
    (void)lds_cfg;

    plstm_forecast_kernel<<<B_TOT / 16, 128, LDS_TOTAL, stream>>>(
        inputs, b1, tau1, shift1, ron1, b2, tau2, shift2, ron2, bd1, bo, wpk, out);
}
